// Attention_2491081031946
// MI455X (gfx1250) — matrix operand order repaired and checked by reading, not compiled, not run on hardware
//
#include <hip/hip_runtime.h>
#include <hip/hip_bf16.h>

// ---------------------------------------------------------------------------
// Attention forward for MI455X (gfx1250, wave32, WMMA + async/TDM staging).
//   B=4, S=2048, D_MODEL=1024, H=16, D_HEAD=64
//   Kernel 1 (qkv_gemm): X @ W^T + b -> bf16 Q[BH,S,64], K[BH,S,64], Vt[BH,64,S]
//   Kernel 2 (flash_attn): online-softmax flash attention, 32-key tiles,
//       K tile staged by TDM tensor_load_to_lds, V tile by async-to-LDS b128
//       (inline asm), double buffered; S=Q*K^T and O+=P*V via
//       v_wmma_f32_16x16x32_bf16.
// ---------------------------------------------------------------------------

typedef __bf16 bf16_t;
typedef bf16_t v16bf __attribute__((ext_vector_type(16)));
typedef float  v8f   __attribute__((ext_vector_type(8)));
typedef unsigned int u32x4 __attribute__((ext_vector_type(4)));
typedef int          i32x4 __attribute__((ext_vector_type(4)));
typedef int          i32x8 __attribute__((ext_vector_type(8)));

union FragU {
    v16bf v;
    uint4 q[2];
};

#define S_LEN   2048
#define DMODEL  1024
#define NHEAD   16
#define DHEAD   64
#define LDSTR   48
#define NTILES  (S_LEN / 32)

// ---- CDNA5 feature paths -------------------------------------------------
#if defined(__gfx1250__) && __has_builtin(__builtin_amdgcn_tensor_load_to_lds)
#define HAS_TDM 1
#else
#define HAS_TDM 0
#endif

#if HAS_TDM
#define ASYNC_PER_TILE 2   // only V tile via per-lane async (2 instr/wave/tile)
#else
#define ASYNC_PER_TILE 4   // K + V both via per-lane async
#endif

#if defined(__gfx1250__)
#define WAIT_ASYNC(n) asm volatile("s_wait_asynccnt %0" :: "n"(n) : "memory")
#else
#define WAIT_ASYNC(n) do {} while (0)
#endif

// Per-lane async copy of 16 bytes global -> LDS (ASYNCcnt-tracked, no VGPR
// data). GV-mode VGLOBAL syntax: global_load_async_to_lds_b128 vDst, vAddr, off
static __device__ __forceinline__ void async_copy16(const bf16_t* g, bf16_t* l) {
#if defined(__gfx1250__)
    asm volatile("global_load_async_to_lds_b128 %0, %1, off"
                 :: "v"((unsigned)(size_t)l),            // LDS byte address
                    "v"((unsigned long long)(size_t)g)   // 64-bit global address
                 : "memory");
#else
    *(uint4*)l = *(const uint4*)g;   // host-pass / non-gfx1250 fallback
#endif
}

static __device__ __forceinline__ v8f v8f_zero() {
    v8f z = {0.f, 0.f, 0.f, 0.f, 0.f, 0.f, 0.f, 0.f};
    return z;
}

// ---------------------------------------------------------------------------
// Kernel 1: QKV projection GEMM.  grid = (N/128, M/128, 3), block = 256.
// ---------------------------------------------------------------------------
__global__ __launch_bounds__(256) void qkv_gemm_kernel(
    const float* __restrict__ xq, const float* __restrict__ xk,
    const float* __restrict__ xv,
    const float* __restrict__ Wq, const float* __restrict__ Wk,
    const float* __restrict__ Wv,
    const float* __restrict__ bq, const float* __restrict__ bk,
    const float* __restrict__ bv,
    bf16_t* __restrict__ Qo, bf16_t* __restrict__ Ko, bf16_t* __restrict__ Vo)
{
    const int z = blockIdx.z;
    const float* X  = (z == 0) ? xq : ((z == 1) ? xk : xv);
    const float* W  = (z == 0) ? Wq : ((z == 1) ? Wk : Wv);
    const float* Bi = (z == 0) ? bq : ((z == 1) ? bk : bv);

    const int mBase = blockIdx.y * 128;
    const int nBase = blockIdx.x * 128;

    __shared__ bf16_t As[128 * LDSTR];
    __shared__ bf16_t Bs[128 * LDSTR];

    const int tid  = threadIdx.x;
    const int lane = tid & 31;
    const int wv   = tid >> 5;
    const int wm   = wv & 3;
    const int wn   = wv >> 2;
    const int ln   = lane & 15;
    const int hf   = lane >> 4;

    v8f acc[2][4];
#pragma unroll
    for (int i = 0; i < 2; ++i)
#pragma unroll
        for (int j = 0; j < 4; ++j) acc[i][j] = v8f_zero();

    for (int k0 = 0; k0 < DMODEL; k0 += 32) {
        __syncthreads();

        // --- staging: issue all 8 tile loads first (one clause, counted
        // waits), then convert fp32->bf16 and store to LDS. ---
        float4 fa[4], fb[4];
        const float* Xp = X + (size_t)mBase * DMODEL + k0;
        const float* Wp = W + (size_t)nBase * DMODEL + k0;
#pragma unroll
        for (int it = 0; it < 4; ++it) {
            const int idx  = tid + it * 256;
            const int row  = idx >> 3;
            const int col4 = idx & 7;
            fa[it] = ((const float4*)(Xp + (size_t)row * DMODEL))[col4];
            fb[it] = ((const float4*)(Wp + (size_t)row * DMODEL))[col4];
        }
#pragma unroll
        for (int it = 0; it < 4; ++it) {
            const int idx  = tid + it * 256;
            const int row  = idx >> 3;
            const int col4 = idx & 7;
            union { bf16_t h[4]; uint2 u; } pa;
            pa.h[0] = (bf16_t)fa[it].x; pa.h[1] = (bf16_t)fa[it].y;
            pa.h[2] = (bf16_t)fa[it].z; pa.h[3] = (bf16_t)fa[it].w;
            *(uint2*)(&As[row * LDSTR + col4 * 4]) = pa.u;
            union { bf16_t h[4]; uint2 u; } pb;
            pb.h[0] = (bf16_t)fb[it].x; pb.h[1] = (bf16_t)fb[it].y;
            pb.h[2] = (bf16_t)fb[it].z; pb.h[3] = (bf16_t)fb[it].w;
            *(uint2*)(&Bs[row * LDSTR + col4 * 4]) = pb.u;
        }
        __syncthreads();

        // Prefetch next k-tile into cache while WMMAs run (global_prefetch_b8).
        if (k0 + 32 < DMODEL) {
            const int prow = tid >> 1;
            const int pcol = (tid & 1) * 16;
            __builtin_prefetch(X + (size_t)(mBase + prow) * DMODEL + (k0 + 32) + pcol, 0, 1);
            __builtin_prefetch(W + (size_t)(nBase + prow) * DMODEL + (k0 + 32) + pcol, 0, 1);
        }

        FragU a[2], b[4];
        const int alo = hf ? 8 : 0;
#pragma unroll
        for (int i = 0; i < 2; ++i) {
            const bf16_t* p = &As[(wm * 32 + i * 16 + ln) * LDSTR];
            a[i].q[0] = *(const uint4*)(p + alo);
            a[i].q[1] = *(const uint4*)(p + alo + 16);
        }
        const int bo = hf ? 8 : 0;
#pragma unroll
        for (int j = 0; j < 4; ++j) {
            const bf16_t* p = &Bs[(wn * 64 + j * 16 + ln) * LDSTR + bo];
            b[j].q[0] = *(const uint4*)(p);
            b[j].q[1] = *(const uint4*)(p + 16);
        }
#pragma unroll
        for (int i = 0; i < 2; ++i)
#pragma unroll
            for (int j = 0; j < 4; ++j)
                acc[i][j] = __builtin_amdgcn_wmma_f32_16x16x32_bf16(
                    false, a[i].v, false, b[j].v, (short)0, acc[i][j], false, false);
    }

#pragma unroll
    for (int i = 0; i < 2; ++i) {
#pragma unroll
        for (int j = 0; j < 4; ++j) {
            const int n    = nBase + wn * 64 + j * 16 + ln;
            const float bb = Bi[n];
            const int h    = n >> 6;
            const int d    = n & 63;
#pragma unroll
            for (int e = 0; e < 8; ++e) {
                const int m    = mBase + wm * 32 + i * 16 + hf * 8 + e;
                const int bidx = m >> 11;
                const int s    = m & (S_LEN - 1);
                const size_t bh = (size_t)(bidx * NHEAD + h);
                const bf16_t val = (bf16_t)(acc[i][j][e] + bb);
                if (z == 0)      Qo[(bh * S_LEN + s) * DHEAD + d] = val;
                else if (z == 1) Ko[(bh * S_LEN + s) * DHEAD + d] = val;
                else             Vo[(bh * DHEAD + d) * S_LEN + s] = val;
            }
        }
    }
}

// ---------------------------------------------------------------------------
// Kernel 2: flash attention.  grid = (S/64, B*H), block = 128 (4 waves).
// Double-buffered K/V staging: K via TDM tensor_load_to_lds (wave 0 issues,
// TENSORcnt), V via global_load_async_to_lds_b128 (ASYNCcnt), overlapped
// with WMMA compute on the other buffer.
// ---------------------------------------------------------------------------
__global__ __launch_bounds__(128) void flash_attn_kernel(
    const bf16_t* __restrict__ Q,   // [BH, S, 64]
    const bf16_t* __restrict__ K,   // [BH, S, 64]
    const bf16_t* __restrict__ Vt,  // [BH, 64, S]
    const unsigned char* __restrict__ mask,  // [B, S, S] bool
    float* __restrict__ out)        // [B, S, 1024]
{
    const int bh    = blockIdx.y;
    const int b     = bh >> 4;
    const int h     = bh & 15;
    const int qBase = blockIdx.x * 64;

    const int tid  = threadIdx.x;
    const int lane = tid & 31;
    const int wv   = tid >> 5;
    const int ln   = lane & 15;
    const int hf   = lane >> 4;

    __shared__ bf16_t Ks[2][32 * 64];    // [key][d]
    __shared__ bf16_t Vs[2][64 * 32];    // [d][key]
    __shared__ bf16_t Ps[4][16 * 32];    // per-wave P staging

    // Stage one 32-key tile (K: 32x64, V^T: 64x32) into buffer `buf`.
    auto stage_tile = [&](int kvOff, int buf) {
#if HAS_TDM
        if (wv == 0) {
            // D# for a 2D tile: 32 rows x 64 elems, 2B elems, row stride 64.
            const unsigned ldsOff = (unsigned)(size_t)&Ks[buf][0];
            const unsigned long long ga =
                (unsigned long long)(size_t)(K + ((size_t)bh * S_LEN + kvOff) * DHEAD);
            const u32x4 g0 = { 1u,                               // count=1 user D#
                               ldsOff,                           // lds_addr
                               (unsigned)ga,                     // global_addr lo
                               ((unsigned)((ga >> 32) & 0x1FFFFFFu)) | (2u << 30) }; // hi|type=2
            const i32x8 g1 = { 0x00010000,        // data_size=1 (2 bytes)
                               0x00400000,        // tensor_dim0 = 64 (lo16<<16)
                               0x00200000,        // tensor_dim1 = 32 (lo16<<16)
                               0x00400000,        // tile_dim0   = 64 (<<16)
                               32,                // tile_dim1=32, tile_dim2=0
                               64,                // tensor_dim0_stride = 64
                               0, 0 };
            const i32x4 gz = { 0, 0, 0, 0 };
#if __clang_major__ >= 23
            const i32x8 g4 = { 0, 0, 0, 0, 0, 0, 0, 0 };
            __builtin_amdgcn_tensor_load_to_lds(g0, g1, gz, gz, g4, 0);
#else
            __builtin_amdgcn_tensor_load_to_lds(g0, g1, gz, gz, 0);
#endif
        }
#else
#pragma unroll
        for (int it = 0; it < 2; ++it) {
            const int c  = tid + it * 128;
            const int kr = c >> 3, kc = c & 7;
            async_copy16(K + ((size_t)bh * S_LEN + kvOff + kr) * DHEAD + kc * 8,
                         &Ks[buf][kr * 64 + kc * 8]);
        }
#endif
#pragma unroll
        for (int it = 0; it < 2; ++it) {
            const int c  = tid + it * 128;
            const int vr = c >> 2, vc = c & 3;
            async_copy16(Vt + ((size_t)bh * DHEAD + vr) * S_LEN + kvOff + vc * 8,
                         &Vs[buf][vr * 32 + vc * 8]);
        }
    };

    // Q fragments for this wave's 16 rows, resident in VGPRs all kernel.
    FragU aq[2];
    {
        const int row = qBase + wv * 16 + ln;
        const bf16_t* qp = Q + ((size_t)bh * S_LEN + row) * DHEAD;
        const int alo = hf ? 8 : 0;
#pragma unroll
        for (int kd = 0; kd < 2; ++kd) {
            aq[kd].q[0] = *(const uint4*)(qp + kd * 32 + alo);
            aq[kd].q[1] = *(const uint4*)(qp + kd * 32 + alo + 16);
        }
    }

    float rm[8], rs[8];
#pragma unroll
    for (int e = 0; e < 8; ++e) { rm[e] = -3.0e38f; rs[e] = 0.0f; }
    v8f o[4];
#pragma unroll
    for (int j = 0; j < 4; ++j) o[j] = v8f_zero();

    const float scale = 0.125f;   // 1/sqrt(64)

    stage_tile(0, 0);             // prologue: tile 0 in flight

    for (int t = 0; t < NTILES; ++t) {
        const int  cur  = t & 1;
        const bool more = (t + 1 < NTILES);

        if (more) stage_tile((t + 1) * 32, cur ^ 1);   // issue next tile

        if (more) { WAIT_ASYNC(ASYNC_PER_TILE); } else { WAIT_ASYNC(0); }
#if HAS_TDM
        if (wv == 0) {
            if (more) __builtin_amdgcn_s_wait_tensorcnt(1);
            else      __builtin_amdgcn_s_wait_tensorcnt(0);
        }
#endif
        __syncthreads();          // current tile visible to all waves

        // ---- scores: 16 rows x 32 keys ----
        v8f sc[2];
        sc[0] = v8f_zero(); sc[1] = v8f_zero();
#pragma unroll
        for (int kd = 0; kd < 2; ++kd) {
#pragma unroll
            for (int j = 0; j < 2; ++j) {
                FragU bk;
                const bf16_t* p = &Ks[cur][(j * 16 + ln) * 64 + kd * 32 + hf * 8];
                bk.q[0] = *(const uint4*)(p);
                bk.q[1] = *(const uint4*)(p + 16);
                sc[j] = __builtin_amdgcn_wmma_f32_16x16x32_bf16(
                    false, aq[kd].v, false, bk.v, (short)0, sc[j], false, false);
            }
        }

        // ---- scale + mask ----
        const int kv    = t * 32;
        const int rowLo = qBase + wv * 16 + hf * 8;
#pragma unroll
        for (int j = 0; j < 2; ++j) {
            const int key = kv + j * 16 + ln;
#pragma unroll
            for (int e = 0; e < 8; ++e) {
                const unsigned char mk =
                    mask[((size_t)b * S_LEN + (rowLo + e)) * S_LEN + key];
                const float s = sc[j][e] * scale;
                sc[j][e] = mk ? -1.0e7f : s;
            }
        }

        // ---- online softmax ----
#pragma unroll
        for (int e = 0; e < 8; ++e) {
            float mx = fmaxf(sc[0][e], sc[1][e]);
            mx = fmaxf(mx, __shfl_xor(mx, 1, 32));
            mx = fmaxf(mx, __shfl_xor(mx, 2, 32));
            mx = fmaxf(mx, __shfl_xor(mx, 4, 32));
            mx = fmaxf(mx, __shfl_xor(mx, 8, 32));
            const float mnew = fmaxf(rm[e], mx);
            const float corr = __expf(rm[e] - mnew);
            rm[e] = mnew;
            const float p0 = __expf(sc[0][e] - mnew);
            const float p1 = __expf(sc[1][e] - mnew);
            sc[0][e] = p0; sc[1][e] = p1;
            float ps = p0 + p1;
            ps += __shfl_xor(ps, 1, 32);
            ps += __shfl_xor(ps, 2, 32);
            ps += __shfl_xor(ps, 4, 32);
            ps += __shfl_xor(ps, 8, 32);
            rs[e] = rs[e] * corr + ps;
#pragma unroll
            for (int j = 0; j < 4; ++j) o[j][e] *= corr;
        }

        // ---- P -> wave-private LDS staging (C-layout -> A-layout) ----
#pragma unroll
        for (int j = 0; j < 2; ++j)
#pragma unroll
            for (int e = 0; e < 8; ++e)
                Ps[wv][(hf * 8 + e) * 32 + j * 16 + ln] = (bf16_t)sc[j][e];

        asm volatile("s_wait_dscnt 0" ::: "memory");

        // ---- O += P @ V ----
        FragU ap;
        {
            const int alo = hf ? 8 : 0;
            const bf16_t* p = &Ps[wv][ln * 32];
            ap.q[0] = *(const uint4*)(p + alo);
            ap.q[1] = *(const uint4*)(p + alo + 16);
        }
#pragma unroll
        for (int j = 0; j < 4; ++j) {
            FragU bvf;
            const bf16_t* p = &Vs[cur][(j * 16 + ln) * 32 + hf * 8];
            bvf.q[0] = *(const uint4*)(p);
            bvf.q[1] = *(const uint4*)(p + 16);
            o[j] = __builtin_amdgcn_wmma_f32_16x16x32_bf16(
                false, ap.v, false, bvf.v, (short)0, o[j], false, false);
        }

        __syncthreads();          // all reads of buffer `cur` done before reuse
    }

    // ---- epilogue: normalize, write fp32 [B, S, H*D] ----
#pragma unroll
    for (int j = 0; j < 4; ++j) {
        const int d = j * 16 + ln;
#pragma unroll
        for (int e = 0; e < 8; ++e) {
            const int row = qBase + wv * 16 + hf * 8 + e;
            out[((size_t)b * S_LEN + row) * DMODEL + h * DHEAD + d] = o[j][e] / rs[e];
        }
    }
}

// ---------------------------------------------------------------------------
extern "C" void kernel_launch(void* const* d_in, const int* in_sizes, int n_in,
                              void* d_out, int out_size, void* d_ws, size_t ws_size,
                              hipStream_t stream) {
    (void)in_sizes; (void)n_in; (void)out_size; (void)ws_size;
    const float* q  = (const float*)d_in[0];
    const float* k  = (const float*)d_in[1];
    const float* v  = (const float*)d_in[2];
    const unsigned char* mask = (const unsigned char*)d_in[3];
    const float* Wq = (const float*)d_in[4];
    const float* bq = (const float*)d_in[5];
    const float* Wk = (const float*)d_in[6];
    const float* bk = (const float*)d_in[7];
    const float* Wv = (const float*)d_in[8];
    const float* bv = (const float*)d_in[9];

    const size_t perBuf = (size_t)4 * NHEAD * S_LEN * DHEAD;  // 8M elems
    bf16_t* Qh = (bf16_t*)d_ws;
    bf16_t* Kh = Qh + perBuf;
    bf16_t* Vh = Kh + perBuf;

    dim3 gGemm(DMODEL / 128, (4 * S_LEN) / 128, 3);
    qkv_gemm_kernel<<<gGemm, 256, 0, stream>>>(q, k, v, Wq, Wk, Wv, bq, bk, bv,
                                               Qh, Kh, Vh);

    dim3 gAttn(S_LEN / 64, 4 * NHEAD);
    flash_attn_kernel<<<gAttn, 128, 0, stream>>>(Qh, Kh, Vh, mask, (float*)d_out);
}
